// TransformerBlock_1194000908473
// MI455X (gfx1250) — matrix operand order repaired and checked by reading, not compiled, not run on hardware
//
#include <hip/hip_runtime.h>
#include <hip/hip_bf16.h>
#include <math.h>

// Problem constants (match reference)
#define BB   2
#define SS   2048
#define DD   1024
#define HH   16
#define HD   64
#define DFF  4096
#define BS   (BB * SS)   // 4096 tokens

typedef __attribute__((ext_vector_type(16))) _Float16 v16h;
typedef __attribute__((ext_vector_type(8)))  _Float16 v8h;
typedef __attribute__((ext_vector_type(8)))  float    v8f;
typedef __attribute__((ext_vector_type(4)))  int      i32x4;

// ---------------------------------------------------------------------------
// Async global->LDS copy (gfx1250 GLOBAL_LOAD_ASYNC_TO_LDS_B128, ASYNCcnt).
// Builtin signature (from compiler diagnostics): param0 = i32x4 AS1*,
// param1 = i32x4 AS3*, then two immediate ints (offset, cpol).
// Guarded: falls back to a synchronous LDS store if the builtin is absent.
// ---------------------------------------------------------------------------
#if __has_builtin(__builtin_amdgcn_global_load_async_to_lds_b128)
#define ASYNC_COPY16(gsrc, ldst)                                             \
  __builtin_amdgcn_global_load_async_to_lds_b128(                            \
      (__attribute__((address_space(1))) i32x4*)(uintptr_t)(gsrc),           \
      (__attribute__((address_space(3))) i32x4*)(ldst), 0, 0)
#else
#define ASYNC_COPY16(gsrc, ldst)                                             \
  do { *(v8h*)(ldst) = *(const v8h*)(gsrc); } while (0)
#endif

#if __has_builtin(__builtin_amdgcn_s_wait_asynccnt)
#define WAIT_ASYNC() __builtin_amdgcn_s_wait_asynccnt(0)
#elif __has_builtin(__builtin_amdgcn_global_load_async_to_lds_b128)
#define WAIT_ASYNC() asm volatile("s_wait_asynccnt 0" ::: "memory")
#else
#define WAIT_ASYNC() do { } while (0)
#endif

// ---------------------------------------------------------------------------
// WMMA helpers (gfx1250, wave32). A and B both use contiguous k order: k = 16*(lane/16) + element.
// ---------------------------------------------------------------------------
__device__ __forceinline__ v8f wmma32(v16h a, v16h b, v8f c) {
  return __builtin_amdgcn_wmma_f32_16x16x32_f16(
      /*neg_a=*/false, a, /*neg_b=*/false, b,
      /*c_mod=*/(short)0, c, /*reuse_a=*/false, /*reuse_b=*/false);
}

// B fragment (32x16, K x N): lane n%16 holds column n; lanes<16 hold K=0..15,
// lanes>=16 hold K=16..31.  With B stored transposed (N x K row-major) each
// lane reads 16 contiguous halves.
__device__ __forceinline__ v16h load16h(const _Float16* p) {
  union { v16h v; v8h h[2]; } u;
  u.h[0] = *(const v8h*)(p);
  u.h[1] = *(const v8h*)(p + 8);
  return u.v;
}

// A fragment (16x32, M x K), row-major A: lane m%16 holds row m.
// lanes<16: halves = K{0..15}; lanes>=16: halves = K{16..31}.
__device__ __forceinline__ v16h load_a_frag(const _Float16* rowp, int lane) {
  const int off = (lane & 16) ? 16 : 0;
  union { v16h v; v8h h[2]; } u;
  u.h[0] = *(const v8h*)(rowp + off);
  u.h[1] = *(const v8h*)(rowp + off + 8);
  return u.v;
}

// ---------------------------------------------------------------------------
// Weight cast + transpose: W (K x N, f32) -> WT (N x K, f16)
// ---------------------------------------------------------------------------
__global__ __launch_bounds__(256) void cast_transpose_kernel(
    const float* __restrict__ W, _Float16* __restrict__ WT, int K, int N) {
  size_t idx = (size_t)blockIdx.x * 256 + threadIdx.x;
  size_t total = (size_t)K * N;
  if (idx >= total) return;
  int k = (int)(idx / N);
  int n = (int)(idx % N);
  WT[(size_t)n * K + k] = (_Float16)W[idx];
}

// V (BS x D, f16, token-major) -> VT[b][h][d][s] (f16) for contiguous B-frags
__global__ __launch_bounds__(256) void v_transpose_kernel(
    const _Float16* __restrict__ v, _Float16* __restrict__ vt) {
  size_t idx = (size_t)blockIdx.x * 256 + threadIdx.x;
  size_t total = (size_t)BS * DD;
  if (idx >= total) return;
  int tok = (int)(idx / DD);
  int c   = (int)(idx % DD);
  int b = tok / SS, s = tok % SS;
  int h = c / HD,  d = c % HD;
  vt[(((size_t)b * HH + h) * HD + d) * SS + s] = v[idx];
}

// ---------------------------------------------------------------------------
// LayerNorm: one block per token row, f32 in -> f16 out
// ---------------------------------------------------------------------------
__global__ __launch_bounds__(256) void ln_kernel(
    const float* __restrict__ x, const float* __restrict__ scale,
    const float* __restrict__ shift, _Float16* __restrict__ out) {
  const float* xr = x + (size_t)blockIdx.x * DD;
  float s = 0.f, s2 = 0.f;
  for (int i = threadIdx.x; i < DD; i += 256) {
    float v = xr[i]; s += v; s2 += v * v;
  }
  for (int o = 16; o > 0; o >>= 1) {
    s  += __shfl_xor(s,  o, 32);
    s2 += __shfl_xor(s2, o, 32);
  }
  __shared__ float sm[16];
  int wave = threadIdx.x >> 5, lane = threadIdx.x & 31;
  if (lane == 0) { sm[wave] = s; sm[8 + wave] = s2; }
  __syncthreads();
  s = 0.f; s2 = 0.f;
  for (int w = 0; w < 8; ++w) { s += sm[w]; s2 += sm[8 + w]; }
  float mean = s / DD;
  float var  = s2 / DD - mean * mean;
  float rstd = rsqrtf(var + 1e-5f);
  _Float16* orow = out + (size_t)blockIdx.x * DD;
  for (int i = threadIdx.x; i < DD; i += 256)
    orow[i] = (_Float16)(scale[i] * ((xr[i] - mean) * rstd) + shift[i]);
}

// ---------------------------------------------------------------------------
// WMMA GEMM: C = A(f16, MxK row-major) x BT(f16, NxK row-major)^T
// Block: 256 threads = 8 waves; wave w -> rows [mbase, mbase+16), 64 cols.
// B panel (64 x 32 halves = 4KB) is staged in LDS via async load-to-LDS,
// double-buffered so the next panel's DMA overlaps this panel's WMMAs.
// EPI 0: f16 out;  EPI 1: f32 out + bias + residual;  EPI 2: f16 out + bias + gelu
// ---------------------------------------------------------------------------
template <int EPI>
__global__ __launch_bounds__(256) void gemm_kernel(
    const _Float16* __restrict__ A, const _Float16* __restrict__ BT,
    const float* __restrict__ bias, const float* __restrict__ resid,
    void* __restrict__ out, int M, int N, int K) {
  const int tid   = threadIdx.x;
  const int lane  = tid & 31;
  const int wave  = tid >> 5;
  const int mbase = blockIdx.x * 128 + wave * 16;
  const int nbase = blockIdx.y * 64;
  const int nlane = lane & 15;
  const int grp   = lane >> 4;

  // double-buffered B panel: [buf][n(64)][k(32)] halves
  __shared__ _Float16 bsh[2][64 * 32];

  // each of the 256 threads copies one 16-byte chunk per k-step
  const int cp_n = tid >> 2;          // 0..63  (panel row = local n)
  const int cp_c = (tid & 3) * 8;     // 0,8,16,24 (half offset within row)
  const _Float16* cp_src =
      BT + (size_t)(nbase + cp_n) * K + cp_c;

  v8f acc[4] = {};
  const _Float16* arow = A + (size_t)(mbase + nlane) * K;

  const int steps = K >> 5;
  // prologue: stage panel 0
  ASYNC_COPY16(cp_src, &bsh[0][cp_n * 32 + cp_c]);

  for (int i = 0; i < steps; ++i) {
    WAIT_ASYNC();          // this wave's pending panel load has landed
    __syncthreads();       // all waves' chunks visible; prev reads done
    if (i + 1 < steps) {
      ASYNC_COPY16(cp_src + (size_t)(i + 1) * 32,
                   &bsh[(i + 1) & 1][cp_n * 32 + cp_c]);
    }
    const int k0 = i * 32;
    __builtin_prefetch(arow + k0 + 256, 0, 0);
    v16h af = load_a_frag(arow + k0, lane);
    const _Float16* bp = &bsh[i & 1][0];
#pragma unroll
    for (int t = 0; t < 4; ++t) {
      v16h bf = load16h(bp + (t * 16 + nlane) * 32 + grp * 16);
      acc[t] = wmma32(af, bf, acc[t]);
    }
  }

#pragma unroll
  for (int t = 0; t < 4; ++t) {
    const int col = nbase + t * 16 + nlane;
#pragma unroll
    for (int r = 0; r < 8; ++r) {
      const int m = mbase + r + 8 * grp;   // C layout: VGPR r -> row r + 8*grp
      float v = acc[t][r];
      if (EPI == 1) {
        v += bias[col] + resid[(size_t)m * N + col];
        ((float*)out)[(size_t)m * N + col] = v;
      } else if (EPI == 2) {
        v += bias[col];
        float g = 0.5f * v *
            (1.0f + tanhf(0.7978845608028654f * (v + 0.044715f * v * v * v)));
        ((_Float16*)out)[(size_t)m * N + col] = (_Float16)g;
      } else {
        ((_Float16*)out)[(size_t)m * N + col] = (_Float16)v;
      }
    }
  }
  (void)M;
}

// ---------------------------------------------------------------------------
// Flash attention (causal). Grid: (B*H, S/64), block = 128 (4 waves).
// Wave owns 16 q-rows. Per 32-key step: 4 score WMMAs, online softmax,
// P transposed through LDS into an A fragment, 4 ctx WMMAs vs V^T.
// NOTE: waves have different causal trip counts -> no block barriers here;
// all LDS traffic is wave-private (same-wave LDS ops are in order).
// ---------------------------------------------------------------------------
__global__ __launch_bounds__(128) void attention_kernel(
    const _Float16* __restrict__ q, const _Float16* __restrict__ k,
    const _Float16* __restrict__ vt, _Float16* __restrict__ ctx) {
  const int bh   = blockIdx.x;
  const int b    = bh / HH;
  const int h    = bh % HH;
  const int wave = threadIdx.x >> 5;
  const int lane = threadIdx.x & 31;
  const int qbase = blockIdx.y * 64 + wave * 16;
  const int nlane = lane & 15;
  const int grp   = lane >> 4;

  __shared__ float pbuf[4][16 * 32];
  float* pb = pbuf[wave];

  // Q fragments (loop-invariant): head-dim chunks [0,32) and [32,64)
  const _Float16* qrow = q + (size_t)(b * SS + qbase + nlane) * DD + h * HD;
  v16h qf0 = load_a_frag(qrow,      lane);
  v16h qf1 = load_a_frag(qrow + 32, lane);

  v8f acc[4] = {};
  float mrow[8], lrow[8];
#pragma unroll
  for (int r = 0; r < 8; ++r) { mrow[r] = -3.0e38f; lrow[r] = 0.f; }

  const int qlast = qbase + 15;
  for (int kb = 0; kb <= qlast; kb += 32) {
    // ---- scores: 16q x 32k, f16 WMMA over head-dim (K=32 per op) ----
    v8f s[2] = {};
#pragma unroll
    for (int t = 0; t < 2; ++t) {
      const _Float16* krow =
          k + (size_t)(b * SS + kb + t * 16 + nlane) * DD + h * HD + grp * 16;
      v16h kf0 = load16h(krow);
      v16h kf1 = load16h(krow + 32);
      s[t] = wmma32(qf0, kf0, s[t]);
      s[t] = wmma32(qf1, kf1, s[t]);
    }
    // ---- scale + causal mask ----
#pragma unroll
    for (int t = 0; t < 2; ++t) {
      const int key = kb + t * 16 + nlane;
#pragma unroll
      for (int r = 0; r < 8; ++r) {
        float v = s[t][r] * 0.125f;                 // 1/sqrt(64)
        if (key > qbase + r + 8 * grp) v = -3.0e38f;
        s[t][r] = v;
      }
    }
    // ---- online softmax (row reductions across 16-lane groups) ----
    float rm[8];
#pragma unroll
    for (int r = 0; r < 8; ++r) rm[r] = fmaxf(s[0][r], s[1][r]);
#pragma unroll
    for (int o = 1; o < 16; o <<= 1)
#pragma unroll
      for (int r = 0; r < 8; ++r) rm[r] = fmaxf(rm[r], __shfl_xor(rm[r], o, 32));

    float esc[8];
#pragma unroll
    for (int r = 0; r < 8; ++r) {
      float mnew = fmaxf(mrow[r], rm[r]);
      esc[r] = __expf(mrow[r] - mnew);
      mrow[r] = mnew;
    }
    float ps[8] = {0.f, 0.f, 0.f, 0.f, 0.f, 0.f, 0.f, 0.f};
#pragma unroll
    for (int t = 0; t < 2; ++t)
#pragma unroll
      for (int r = 0; r < 8; ++r) {
        float p = __expf(s[t][r] - mrow[r]);
        s[t][r] = p;
        ps[r] += p;
      }
#pragma unroll
    for (int o = 1; o < 16; o <<= 1)
#pragma unroll
      for (int r = 0; r < 8; ++r) ps[r] += __shfl_xor(ps[r], o, 32);
#pragma unroll
    for (int r = 0; r < 8; ++r) lrow[r] = lrow[r] * esc[r] + ps[r];

    // rescale ctx accumulators (C-layout rows match this lane's group)
#pragma unroll
    for (int t = 0; t < 4; ++t)
#pragma unroll
      for (int r = 0; r < 8; ++r) acc[t][r] *= esc[r];

    // ---- transpose P (C layout -> A layout) through wave-private LDS ----
#pragma unroll
    for (int t = 0; t < 2; ++t)
#pragma unroll
      for (int r = 0; r < 8; ++r)
        pb[(r + 8 * grp) * 32 + t * 16 + nlane] = s[t][r];
    asm volatile("s_wait_dscnt 0" ::: "memory");

    union { v16h v; _Float16 e[16]; } pf;
    const int aoff = grp ? 16 : 0;
#pragma unroll
    for (int i = 0; i < 8; ++i) {
      pf.e[i]     = (_Float16)pb[nlane * 32 + aoff + i];
      pf.e[8 + i] = (_Float16)pb[nlane * 32 + aoff + 8 + i];
    }

    // ---- ctx += P(16x32) @ V(32x64), B-frags from V^T are contiguous ----
#pragma unroll
    for (int t = 0; t < 4; ++t) {
      const _Float16* vp =
          vt + (((size_t)bh * HD) + t * 16 + nlane) * SS + kb + grp * 16;
      v16h vf = load16h(vp);
      acc[t] = wmma32(pf.v, vf, acc[t]);
    }
  }

  // ---- epilogue: divide by l, write ctx in [b, s, h*HD + d] layout ----
#pragma unroll
  for (int t = 0; t < 4; ++t)
#pragma unroll
    for (int r = 0; r < 8; ++r) {
      const int m = qbase + r + 8 * grp;
      ctx[(size_t)(b * SS + m) * DD + h * HD + t * 16 + nlane] =
          (_Float16)(acc[t][r] / lrow[r]);
    }
}

// ---------------------------------------------------------------------------
// Host-side launcher
// ---------------------------------------------------------------------------
extern "C" void kernel_launch(void* const* d_in, const int* in_sizes, int n_in,
                              void* d_out, int out_size, void* d_ws, size_t ws_size,
                              hipStream_t stream) {
  const float* x        = (const float*)d_in[0];
  const float* ln_scale = (const float*)d_in[1];
  const float* ln_shift = (const float*)d_in[2];
  const float* Wq       = (const float*)d_in[3];
  const float* Wk       = (const float*)d_in[4];
  const float* Wv       = (const float*)d_in[5];
  const float* Wo       = (const float*)d_in[6];
  const float* bo       = (const float*)d_in[7];
  const float* W1       = (const float*)d_in[8];
  const float* b1       = (const float*)d_in[9];
  const float* W2       = (const float*)d_in[10];
  const float* b2       = (const float*)d_in[11];
  float* out = (float*)d_out;

  // workspace carve-up
  char* w = (char*)d_ws;
  auto alloc = [&](size_t bytes) -> void* {
    void* p = (void*)w;
    w += (bytes + 255) & ~(size_t)255;
    return p;
  };
  _Float16* WqT  = (_Float16*)alloc((size_t)DD * DD * 2);
  _Float16* WkT  = (_Float16*)alloc((size_t)DD * DD * 2);
  _Float16* WvT  = (_Float16*)alloc((size_t)DD * DD * 2);
  _Float16* WoT  = (_Float16*)alloc((size_t)DD * DD * 2);
  _Float16* W1T  = (_Float16*)alloc((size_t)DD * DFF * 2);
  _Float16* W2T  = (_Float16*)alloc((size_t)DFF * DD * 2);
  _Float16* h1   = (_Float16*)alloc((size_t)BS * DD * 2);
  _Float16* qb   = (_Float16*)alloc((size_t)BS * DD * 2);
  _Float16* kb   = (_Float16*)alloc((size_t)BS * DD * 2);
  _Float16* vb   = (_Float16*)alloc((size_t)BS * DD * 2);
  _Float16* vtb  = (_Float16*)alloc((size_t)BS * DD * 2);
  _Float16* ctxb = (_Float16*)alloc((size_t)BS * DD * 2);
  float*    x1   = (float*)   alloc((size_t)BS * DD * 4);
  _Float16* h2   = (_Float16*)alloc((size_t)BS * DD * 2);
  _Float16* a1   = (_Float16*)alloc((size_t)BS * DFF * 2);
  (void)ws_size; (void)n_in; (void)in_sizes; (void)out_size;

  const int TPB = 256;
  auto eb = [](size_t total) { return (unsigned)((total + 255) / 256); };

  // 1) weights -> f16 transposed
  cast_transpose_kernel<<<eb((size_t)DD * DD), TPB, 0, stream>>>(Wq, WqT, DD, DD);
  cast_transpose_kernel<<<eb((size_t)DD * DD), TPB, 0, stream>>>(Wk, WkT, DD, DD);
  cast_transpose_kernel<<<eb((size_t)DD * DD), TPB, 0, stream>>>(Wv, WvT, DD, DD);
  cast_transpose_kernel<<<eb((size_t)DD * DD), TPB, 0, stream>>>(Wo, WoT, DD, DD);
  cast_transpose_kernel<<<eb((size_t)DD * DFF), TPB, 0, stream>>>(W1, W1T, DD, DFF);
  cast_transpose_kernel<<<eb((size_t)DFF * DD), TPB, 0, stream>>>(W2, W2T, DFF, DD);

  // 2) LN1
  ln_kernel<<<BS, TPB, 0, stream>>>(x, ln_scale, ln_shift, h1);

  // 3) QKV projections (f16 out)
  dim3 gQKV(BS / 128, DD / 64);
  gemm_kernel<0><<<gQKV, TPB, 0, stream>>>(h1, WqT, nullptr, nullptr, qb, BS, DD, DD);
  gemm_kernel<0><<<gQKV, TPB, 0, stream>>>(h1, WkT, nullptr, nullptr, kb, BS, DD, DD);
  gemm_kernel<0><<<gQKV, TPB, 0, stream>>>(h1, WvT, nullptr, nullptr, vb, BS, DD, DD);

  // 4) V -> V^T per (b,h)
  v_transpose_kernel<<<eb((size_t)BS * DD), TPB, 0, stream>>>(vb, vtb);

  // 5) causal flash attention
  dim3 gAtt(BB * HH, SS / 64);
  attention_kernel<<<gAtt, 128, 0, stream>>>(qb, kb, vtb, ctxb);

  // 6) output projection + bias + residual -> x1 (f32)
  gemm_kernel<1><<<gQKV, TPB, 0, stream>>>(ctxb, WoT, bo, x, x1, BS, DD, DD);

  // 7) LN2 (reuses LN1 params, as the reference does)
  ln_kernel<<<BS, TPB, 0, stream>>>(x1, ln_scale, ln_shift, h2);

  // 8) FFN up + GELU (f16 out)
  dim3 gFF1(BS / 128, DFF / 64);
  gemm_kernel<2><<<gFF1, TPB, 0, stream>>>(h2, W1T, b1, nullptr, a1, BS, DFF, DD);

  // 9) FFN down + bias + residual -> out (f32)
  gemm_kernel<1><<<gQKV, TPB, 0, stream>>>(a1, W2T, b2, x1, out, BS, DD, DFF);
}
